// ScaledDotProductAttention_36971078484007
// MI455X (gfx1250) — matrix operand order repaired and checked by reading, not compiled, not run on hardware
//
#include <hip/hip_runtime.h>

// CDNA5 / gfx1250 flash-style causal attention that also materializes the
// full post-softmax attention matrix (reference returns (output, attn)).
//
// One wave32 owns a 16-row query tile. bf16 WMMA 16x16x32 for both QK^T and PV.
// Softmax is computed in base-2 (log2e folded into the Q prescale) so every
// exponential is a bare v_exp_f32 with no preceding multiply
// (__builtin_amdgcn_exp2f).
// Pass A: per-lane online max/sum (no cross-lane ops in the loop), single
//         butterfly reduction at the end.
// Pass B: recompute scores, normalize, store attn, accumulate O = P*V.

typedef __attribute__((ext_vector_type(16))) __bf16 v16bf;
typedef __attribute__((ext_vector_type(8)))  float  v8f;

#define SEQ   2048
#define HD    64
#define NTILE (SEQ / 16)   // 128 query tiles per (b,h)
#define BH    32           // B*H

__global__ __launch_bounds__(128) void fa_causal_attn_kernel(
    const float* __restrict__ qg, const float* __restrict__ kg,
    const float* __restrict__ vg, float* __restrict__ outg,
    float* __restrict__ attng)
{
    __shared__ __align__(16) __bf16 Pbuf[4][16][32];   // per-wave P staging (bf16)

    const int wave = threadIdx.x >> 5;
    const int lane = threadIdx.x & 31;
    const int n    = lane & 15;     // column / N index within 16
    const int hi   = lane >> 4;     // half-wave select

    const int tile = blockIdx.x * 4 + wave;          // global 16-row query tile
    const int qt   = tile & (NTILE - 1);             // tile index within (b,h)
    const int bh   = tile >> 7;                      // which (b,h)

    const float* Q = qg + (size_t)bh * SEQ * HD;
    const float* K = kg + (size_t)bh * SEQ * HD;
    const float* V = vg + (size_t)bh * SEQ * HD;
    float* O = outg  + (size_t)bh * SEQ * HD;
    float* A = attng + (size_t)bh * SEQ * SEQ;
    const int qr = qt * 16;                          // first query row of tile

    // rsqrt(512) * log2(e): scores live in the exp2 domain -> bare v_exp_f32.
    constexpr float SCALE = 0.04419417382415922f * 1.4426950408889634f;

    // ---- Q A-fragments (prescaled, bf16), contiguous-k (k = 16hi + s) A 16x32 layout ----
    // lane holds row M = n; hi selects K offsets 16hi..16hi+15 within each 32-chunk.
    v16bf qa0, qa1;
    {
        const float* qp = Q + (size_t)(qr + n) * HD + 16 * hi;
        #pragma unroll
        for (int j = 0; j < 8; ++j) {
            qa0[j]     = (__bf16)(qp[j]      * SCALE);   // d = 16hi + j
            qa0[j + 8] = (__bf16)(qp[j + 8]  * SCALE);   // d = 16hi + 8 + j
            qa1[j]     = (__bf16)(qp[j + 32] * SCALE);   // d = 32 + 16hi + j
            qa1[j + 8] = (__bf16)(qp[j + 40] * SCALE);   // d = 40 + 16hi + j
        }
    }

    // 16x16 score tile vs key-column tile ct: two K=32 WMMAs over D=64.
    auto score_tile = [&](int ct) -> v8f {
        v8f s = {};
        const float* kp = K + (size_t)(ct * 16 + n) * HD + 16 * hi;
        v16bf b0, b1;                                  // B 32x16: lane = N, hi -> K+16
        #pragma unroll
        for (int j = 0; j < 16; ++j) b0[j] = (__bf16)kp[j];        // d = 16hi + j
        #pragma unroll
        for (int j = 0; j < 16; ++j) b1[j] = (__bf16)kp[j + 32];   // d = 32 + 16hi + j
        s = __builtin_amdgcn_wmma_f32_16x16x32_bf16(false, qa0, false, b0,
                                                    (short)0, s, false, false);
        s = __builtin_amdgcn_wmma_f32_16x16x32_bf16(false, qa1, false, b1,
                                                    (short)0, s, false, false);
        return s;
    };

    // ---- Pass A: per-lane online softmax statistics (no cross-lane ops) ----
    // C-layout row mapping: slot r0 <-> row qr + r0 + 8*hi, col = ct*16 + n.
    // Each lane tracks max/sum over ITS column subset; one butterfly at the end.
    float m[8], l[8];
    #pragma unroll
    for (int r0 = 0; r0 < 8; ++r0) { m[r0] = -1e30f; l[r0] = 0.0f; }

    for (int ct = 0; ct <= qt; ++ct) {
        if (ct < qt)   // next K tile is L2-resident; nudge it toward the WGP
            __builtin_prefetch(K + (size_t)((ct + 1) * 16 + n) * HD + 16 * hi, 0, 0);
        v8f s = score_tile(ct);
        const int col = ct * 16 + n;
        #pragma unroll
        for (int r0 = 0; r0 < 8; ++r0) {
            const int row = qr + r0 + 8 * hi;
            const float vv = (col <= row) ? s[r0] : -1e30f;   // causal mask
            const float mo = m[r0];
            const float mn = fmaxf(mo, vv);
            // one exp2 per element: t = 2^(min - max); merge via select + fma
            const float t  = __builtin_amdgcn_exp2f(fminf(mo, vv) - mn);
            const bool  gt = vv > mo;
            l[r0] = fmaf(l[r0], gt ? t : 1.0f, gt ? 1.0f : t);
            m[r0] = mn;
        }
    }

    // ---- One-time 16-lane butterfly reduction per row slot ----
    // Fully-masked lanes carry m=-1e30 -> 2^(m_lane - m_row) == 0 kills them.
    float invl[8];
    #pragma unroll
    for (int r0 = 0; r0 < 8; ++r0) {
        const float ml = m[r0];
        float mr = ml;
        mr = fmaxf(mr, __shfl_xor(mr, 1, 16));
        mr = fmaxf(mr, __shfl_xor(mr, 2, 16));
        mr = fmaxf(mr, __shfl_xor(mr, 4, 16));
        mr = fmaxf(mr, __shfl_xor(mr, 8, 16));
        float ls = l[r0] * __builtin_amdgcn_exp2f(ml - mr);
        ls += __shfl_xor(ls, 1, 16);
        ls += __shfl_xor(ls, 2, 16);
        ls += __shfl_xor(ls, 4, 16);
        ls += __shfl_xor(ls, 8, 16);
        m[r0]    = mr;            // row max (log2 domain), broadcast to 16 lanes
        invl[r0] = 1.0f / ls;     // row 1/sum, broadcast to 16 lanes
    }

    // ---- Pass B: recompute, normalize, store attn, accumulate O = P * V ----
    v8f o[4] = {v8f{}, v8f{}, v8f{}, v8f{}};
    __bf16 (*P)[32] = Pbuf[wave];

    for (int ct0 = 0; ct0 <= qt; ct0 += 2) {                // K=32 PV chunks
        #pragma unroll
        for (int par = 0; par < 2; ++par) {
            const int ct = ct0 + par;
            if (ct <= qt) {
                v8f s = score_tile(ct);
                const int col = ct * 16 + n;
                #pragma unroll
                for (int r0 = 0; r0 < 8; ++r0) {
                    const int row = qr + r0 + 8 * hi;
                    const float vv = (col <= row) ? s[r0] : -1e30f;
                    const float p  = __builtin_amdgcn_exp2f(vv - m[r0]) * invl[r0];
                    A[(size_t)row * SEQ + col] = p;          // attn output
                    P[r0 + 8 * hi][par * 16 + n] = (__bf16)p;
                }
            } else {                                         // odd tail: zero-pad
                #pragma unroll
                for (int r0 = 0; r0 < 8; ++r0)
                    P[r0 + 8 * hi][par * 16 + n] = (__bf16)0.0f;
            }
        }

        // Same-wave DS RAW is in-order per ISA; explicit wait for safety.
        asm volatile("s_wait_dscnt 0" ::: "memory");

        // Re-read P in WMMA A layout (16x32 bf16): lane row = n, hi -> K offsets.
        v16bf pa;
        {
            uint4 g0 = *(const uint4*)&P[n][16 * hi];        // K = 16hi .. 16hi+7
            uint4 g1 = *(const uint4*)&P[n][16 * hi + 8];    // K = 16hi+8 .. 16hi+15
            __builtin_memcpy(&pa, &g0, 16);
            __builtin_memcpy((char*)&pa + 16, &g1, 16);
        }

        // V as B 32x16 per 16-wide d-slice; accumulate four 16x16 O tiles.
        const int s0 = ct0 * 16;
        #pragma unroll
        for (int nt = 0; nt < 4; ++nt) {
            v16bf vb;
            const float* vp = V + (size_t)(s0 + 16 * hi) * HD + nt * 16 + n;
            #pragma unroll
            for (int j = 0; j < 16; ++j)
                vb[j] = (__bf16)vp[(size_t)j * HD];          // V[s0+16hi+j][nt*16+n]
            o[nt] = __builtin_amdgcn_wmma_f32_16x16x32_bf16(false, pa, false, vb,
                                                            (short)0, o[nt], false, false);
        }
    }

    // ---- Write O (16 x 64) ----
    #pragma unroll
    for (int nt = 0; nt < 4; ++nt)
        #pragma unroll
        for (int r0 = 0; r0 < 8; ++r0)
            O[(size_t)(qr + r0 + 8 * hi) * HD + nt * 16 + n] = o[nt][r0];

    // ---- Zero-fill the fully-masked upper-triangle tiles of attn ----
    // (harness poisons d_out; reference softmax yields exact zeros there)
    {
        const int c0 = (qt + 1) * 16;                        // first untouched col
        const float4 z4 = make_float4(0.f, 0.f, 0.f, 0.f);
        for (int r = 0; r < 16; ++r) {
            float4* rowp = (float4*)(A + (size_t)(qr + r) * SEQ);
            for (int cc = (c0 >> 2) + lane; cc < (SEQ >> 2); cc += 32)
                rowp[cc] = z4;
        }
    }
}

extern "C" void kernel_launch(void* const* d_in, const int* in_sizes, int n_in,
                              void* d_out, int out_size, void* d_ws, size_t ws_size,
                              hipStream_t stream) {
    const float* q = (const float*)d_in[0];
    const float* k = (const float*)d_in[1];
    const float* v = (const float*)d_in[2];
    // d_in[3] is the causal mask; causality is applied analytically in-kernel.
    float* outp = (float*)d_out;                             // (B,H,S,D) fp32
    float* attn = outp + (size_t)BH * SEQ * HD;              // (B,H,S,S) fp32

    const int total_tiles = BH * NTILE;                      // 4096 wave-tiles
    dim3 grid(total_tiles / 4);                              // 4 waves / block
    fa_causal_attn_kernel<<<grid, 128, 0, stream>>>(q, k, v, outp, attn);
}
